// NodeMultiHeadAttention_9414568312871
// MI455X (gfx1250) — matrix operand order repaired and checked by reading, not compiled, not run on hardware
//
#include <hip/hip_runtime.h>
#include <hip/hip_bf16.h>

#define N_TOK 2048
#define HID   256
#define NHEAD 8
#define DHEAD 32

typedef __attribute__((ext_vector_type(16))) __bf16 v16bf;
typedef __attribute__((ext_vector_type(8)))  __bf16 v8bf;
typedef __attribute__((ext_vector_type(8)))  float  v8f;

static __device__ __forceinline__ v16bf cat8(v8bf lo, v8bf hi) {
  return __builtin_shufflevector(lo, hi, 0,1,2,3,4,5,6,7,8,9,10,11,12,13,14,15);
}

// ---------------------------------------------------------------------------
// Bias MLP collapse: both MLPs are affine (no nonlinearity), so
//   bias(i,j) = c0*dist + c1*contact + c2*label + c3
__global__ void nmha_coef_kernel(const float* Wd1, const float* bd1,
                                 const float* Wd2, const float* bd2,
                                 const float* Wc1, const float* bc1,
                                 const float* Wc2, const float* bc2,
                                 float* coef) {
  if (threadIdx.x == 0 && blockIdx.x == 0) {
    float ad = 0.f, bd = 0.f, gd = 0.f, ac = 0.f, bc = 0.f, gc = 0.f;
    for (int h = 0; h < 16; ++h) {
      ad += Wd1[h]      * Wd2[h];
      bd += Wd1[16 + h] * Wd2[h];
      gd += bd1[h]      * Wd2[h];
      ac += Wc1[h]      * Wc2[h];
      bc += Wc1[16 + h] * Wc2[h];
      gc += bc1[h]      * Wc2[h];
    }
    coef[0] = ad;
    coef[1] = ac;
    coef[2] = bd + bc;
    coef[3] = gd + gc + bd2[0] + bc2[0];
  }
}

// Fused bias plane: 16 MB f32, read once, then serves all 8 heads from L2.
__global__ void nmha_bias_kernel(const float* dist, const float* contact,
                                 const int* label, const float* coef,
                                 float* bias) {
  int i = blockIdx.x * blockDim.x + threadIdx.x;
  if (i < N_TOK * N_TOK) {
    float c0 = coef[0], c1 = coef[1], c2 = coef[2], c3 = coef[3];
    bias[i] = fmaf(c0, dist[i], fmaf(c1, contact[i], fmaf(c2, (float)label[i], c3)));
  }
}

__global__ void nmha_cvt_kernel(const float* src, __bf16* dst, int n) {
  int i = blockIdx.x * blockDim.x + threadIdx.x;
  if (i < n) dst[i] = (__bf16)src[i];
}

// W[k][c] (row-major 256x256 f32)  ->  Wt[c][k] bf16, so B-fragments are
// 16 contiguous halves per lane.
__global__ void nmha_wt_kernel(const float* W, __bf16* Wt) {
  int i = blockIdx.x * blockDim.x + threadIdx.x;
  if (i < HID * HID) {
    int c = i >> 8, k = i & 255;
    Wt[(size_t)c * HID + k] = (__bf16)W[(size_t)k * HID + c];
  }
}

// ---------------------------------------------------------------------------
// Generic 16x16-tile WMMA GEMM:  C[M x 256] = A(bf16) @ W + bvec, epilogues:
//   mode 0: store bf16 head layout  dstb[h][n][d]   (q/k; q uses scale)
//   mode 1: store bf16 transposed   dstb[col][n]    (v, per-head V^T)
//   mode 2: store f32               dstf[n][col]    (final output proj)
__global__ __launch_bounds__(256)
void nmha_gemm_kernel(const __bf16* __restrict__ A, const __bf16* __restrict__ Wt,
                      const float* __restrict__ bvec, float scale, int mode,
                      __bf16* __restrict__ dstb, float* __restrict__ dstf) {
  const int lane = threadIdx.x;
  const int t = blockIdx.x * 8 + threadIdx.y;     // tile id, 2048 tiles
  const int tm = t >> 4;                          // 0..127  (rows of 16)
  const int tn = t & 15;                          // 0..15   (cols of 16)
  const int hl = lane & 15;
  const int hb = (lane >> 4) * 8;                 // unused (A half uses ho)
  const int ho = (lane >> 4) * 16;                // B half select

  const __bf16* arow = A  + (size_t)(tm * 16 + hl) * HID;
  const __bf16* wcol = Wt + (size_t)(tn * 16 + hl) * HID;

  v8f acc = {};
#pragma unroll
  for (int kk = 0; kk < HID; kk += 32) {
    v8bf alo = *(const v8bf*)(arow + kk + ho);
    v8bf ahi = *(const v8bf*)(arow + kk + ho + 8);
    v16bf af = cat8(alo, ahi);
    v16bf bf = *(const v16bf*)(wcol + kk + ho);
    acc = __builtin_amdgcn_wmma_f32_16x16x32_bf16(false, af, false, bf,
                                                  (short)0, acc, false, false);
  }

  const int hi8 = (lane >= 16) ? 8 : 0;
#pragma unroll
  for (int r = 0; r < 8; ++r) {
    int orow = tm * 16 + r + hi8;
    int ocol = tn * 16 + hl;
    float v = (acc[r] + bvec[ocol]) * scale;
    if (mode == 0) {
      // head layout: [h][n][32]
      dstb[(size_t)(ocol >> 5) * N_TOK * DHEAD + (size_t)orow * DHEAD + (ocol & 31)] = (__bf16)v;
    } else if (mode == 1) {
      // V^T: [256][N]  (row index == h*32+d)
      dstb[(size_t)ocol * N_TOK + orow] = (__bf16)v;
    } else {
      dstf[(size_t)orow * HID + ocol] = v;
    }
  }
}

// ---------------------------------------------------------------------------
// Flash attention: one wave = 16 query rows of one head, loop over keys in
// blocks of 32.  Per block: 2 WMMA (S = Q K^T), bias add + online softmax,
// P relayout via LDS, 2 WMMA (O += P V).
__global__ __launch_bounds__(256)
void nmha_attn_kernel(const __bf16* __restrict__ qh, const __bf16* __restrict__ kh,
                      const __bf16* __restrict__ vT, const float* __restrict__ bias,
                      __bf16* __restrict__ oh) {
  const int lane = threadIdx.x;
  const int wave = threadIdx.y;
  const int wid  = blockIdx.x * 8 + wave;         // 0..1023
  const int h    = wid >> 7;                      // head
  const int q0   = (wid & 127) * 16;              // query tile base

  __shared__ __align__(16) __bf16 plds_all[8][16 * 32];
  __bf16* plds = plds_all[wave];

  const int hl  = lane & 15;
  const int hb  = (lane >> 4) * 8;
  const int ho  = (lane >> 4) * 16;
  const int hi8 = (lane >= 16) ? 8 : 0;

  // Q A-fragment: held in registers for the whole key loop.
  const __bf16* qrow = qh + ((size_t)h * N_TOK + (q0 + hl)) * DHEAD;
  v16bf afq = cat8(*(const v8bf*)(qrow + ho), *(const v8bf*)(qrow + ho + 8));

  v8f acc0 = {}, acc1 = {};
  float m[8], l[8];
#pragma unroll
  for (int r = 0; r < 8; ++r) { m[r] = -1e30f; l[r] = 0.f; }

  for (int j = 0; j < N_TOK; j += 32) {
    // K^T B-fragments: keys j..j+15 and j+16..j+31 (dims contiguous per lane).
    v16bf bk0 = *(const v16bf*)(kh + ((size_t)h * N_TOK + (j + hl)) * DHEAD + ho);
    v16bf bk1 = *(const v16bf*)(kh + ((size_t)h * N_TOK + (j + 16 + hl)) * DHEAD + ho);
    v8f s0 = {}, s1 = {};
    s0 = __builtin_amdgcn_wmma_f32_16x16x32_bf16(false, afq, false, bk0, (short)0, s0, false, false);
    s1 = __builtin_amdgcn_wmma_f32_16x16x32_bf16(false, afq, false, bk1, (short)0, s1, false, false);

#pragma unroll
    for (int r = 0; r < 8; ++r) {
      int row = r + hi8;
      const float* brow = bias + (size_t)(q0 + row) * N_TOK + j;
      float v0 = s0[r] + brow[hl];
      float v1 = s1[r] + brow[16 + hl];
      // row max across the 16-lane group (masks 1..8 stay within the half)
      float mx = fmaxf(v0, v1);
      mx = fmaxf(mx, __shfl_xor(mx, 1));
      mx = fmaxf(mx, __shfl_xor(mx, 2));
      mx = fmaxf(mx, __shfl_xor(mx, 4));
      mx = fmaxf(mx, __shfl_xor(mx, 8));
      float mn = fmaxf(m[r], mx);
      float alpha = __expf(m[r] - mn);
      float e0 = __expf(v0 - mn);
      float e1 = __expf(v1 - mn);
      float rs = e0 + e1;
      rs += __shfl_xor(rs, 1);
      rs += __shfl_xor(rs, 2);
      rs += __shfl_xor(rs, 4);
      rs += __shfl_xor(rs, 8);
      l[r] = l[r] * alpha + rs;
      m[r] = mn;
      acc0[r] *= alpha;
      acc1[r] *= alpha;
      plds[row * 32 + hl]      = (__bf16)e0;
      plds[row * 32 + 16 + hl] = (__bf16)e1;
    }
    asm volatile("s_wait_dscnt 0" ::: "memory");   // P store -> P load (same wave)

    // P A-fragment (row = hl, keys {ho..ho+15})
    const __bf16* prow = plds + hl * 32;
    v16bf afp = cat8(*(const v8bf*)(prow + ho), *(const v8bf*)(prow + ho + 8));

    // V B-fragments from V^T: dim column, 16 contiguous keys per lane half.
    v16bf bv0 = *(const v16bf*)(vT + (size_t)(h * DHEAD + hl)      * N_TOK + j + ho);
    v16bf bv1 = *(const v16bf*)(vT + (size_t)(h * DHEAD + 16 + hl) * N_TOK + j + ho);
    acc0 = __builtin_amdgcn_wmma_f32_16x16x32_bf16(false, afp, false, bv0, (short)0, acc0, false, false);
    acc1 = __builtin_amdgcn_wmma_f32_16x16x32_bf16(false, afp, false, bv1, (short)0, acc1, false, false);
  }

#pragma unroll
  for (int r = 0; r < 8; ++r) {
    int row = q0 + r + hi8;
    float inv = 1.f / l[r];
    oh[(size_t)row * HID + h * DHEAD + hl]      = (__bf16)(acc0[r] * inv);
    oh[(size_t)row * HID + h * DHEAD + 16 + hl] = (__bf16)(acc1[r] * inv);
  }
}

// ---------------------------------------------------------------------------
extern "C" void kernel_launch(void* const* d_in, const int* in_sizes, int n_in,
                              void* d_out, int out_size, void* d_ws, size_t ws_size,
                              hipStream_t stream) {
  const int*   label   = (const int*)  d_in[0];
  const float* dist    = (const float*)d_in[1];
  const float* contact = (const float*)d_in[2];
  const float* q   = (const float*)d_in[3];
  const float* k   = (const float*)d_in[4];
  const float* v   = (const float*)d_in[5];
  const float* Wq  = (const float*)d_in[6];
  const float* bq  = (const float*)d_in[7];
  const float* Wk  = (const float*)d_in[8];
  const float* bk  = (const float*)d_in[9];
  const float* Wv  = (const float*)d_in[10];
  const float* bv  = (const float*)d_in[11];
  const float* Wo  = (const float*)d_in[12];
  const float* bo  = (const float*)d_in[13];
  const float* Wd1 = (const float*)d_in[14];
  const float* bd1 = (const float*)d_in[15];
  const float* Wd2 = (const float*)d_in[16];
  const float* bd2 = (const float*)d_in[17];
  const float* Wc1 = (const float*)d_in[18];
  const float* bc1 = (const float*)d_in[19];
  const float* Wc2 = (const float*)d_in[20];
  const float* bc2 = (const float*)d_in[21];
  float* out = (float*)d_out;

  // workspace carve-up
  char* ws = (char*)d_ws;
  const size_t NH = (size_t)N_TOK * HID;          // 524288
  __bf16* qbf  = (__bf16*)ws;            ws += NH * 2;
  __bf16* kbf  = (__bf16*)ws;            ws += NH * 2;
  __bf16* vbf  = (__bf16*)ws;            ws += NH * 2;
  __bf16* WTq  = (__bf16*)ws;            ws += (size_t)HID * HID * 2;
  __bf16* WTk  = (__bf16*)ws;            ws += (size_t)HID * HID * 2;
  __bf16* WTv  = (__bf16*)ws;            ws += (size_t)HID * HID * 2;
  __bf16* WTo  = (__bf16*)ws;            ws += (size_t)HID * HID * 2;
  __bf16* qh   = (__bf16*)ws;            ws += NH * 2;
  __bf16* kh   = (__bf16*)ws;            ws += NH * 2;
  __bf16* vT   = (__bf16*)ws;            ws += NH * 2;
  __bf16* oh   = (__bf16*)ws;            ws += NH * 2;
  float*  bias = (float*)ws;             ws += (size_t)N_TOK * N_TOK * 4;
  float*  coef = (float*)ws;             ws += 64;
  (void)ws_size; (void)in_sizes; (void)n_in; (void)out_size;

  // 1) bias coefficients + fused bias plane
  hipLaunchKernelGGL(nmha_coef_kernel, dim3(1), dim3(64), 0, stream,
                     Wd1, bd1, Wd2, bd2, Wc1, bc1, Wc2, bc2, coef);
  hipLaunchKernelGGL(nmha_bias_kernel, dim3((N_TOK * N_TOK + 255) / 256), dim3(256), 0, stream,
                     dist, contact, label, coef, bias);

  // 2) pack activations / weights to bf16
  const int ne = (int)NH;
  hipLaunchKernelGGL(nmha_cvt_kernel, dim3((ne + 255) / 256), dim3(256), 0, stream, q, qbf, ne);
  hipLaunchKernelGGL(nmha_cvt_kernel, dim3((ne + 255) / 256), dim3(256), 0, stream, k, kbf, ne);
  hipLaunchKernelGGL(nmha_cvt_kernel, dim3((ne + 255) / 256), dim3(256), 0, stream, v, vbf, ne);
  hipLaunchKernelGGL(nmha_wt_kernel, dim3(HID * HID / 256), dim3(256), 0, stream, Wq, WTq);
  hipLaunchKernelGGL(nmha_wt_kernel, dim3(HID * HID / 256), dim3(256), 0, stream, Wk, WTk);
  hipLaunchKernelGGL(nmha_wt_kernel, dim3(HID * HID / 256), dim3(256), 0, stream, Wv, WTv);
  hipLaunchKernelGGL(nmha_wt_kernel, dim3(HID * HID / 256), dim3(256), 0, stream, Wo, WTo);

  // 3) projections (2048 tiles of 16x16, 8 waves per block)
  const float scale = 0.17677669529663688f;       // 32^-0.5
  dim3 gblk(32, 8);
  hipLaunchKernelGGL(nmha_gemm_kernel, dim3(256), gblk, 0, stream,
                     qbf, WTq, bq, scale, 0, qh, (float*)nullptr);
  hipLaunchKernelGGL(nmha_gemm_kernel, dim3(256), gblk, 0, stream,
                     kbf, WTk, bk, 1.0f, 0, kh, (float*)nullptr);
  hipLaunchKernelGGL(nmha_gemm_kernel, dim3(256), gblk, 0, stream,
                     vbf, WTv, bv, 1.0f, 1, vT, (float*)nullptr);

  // 4) flash attention: 1024 waves (8 heads x 128 query tiles)
  hipLaunchKernelGGL(nmha_attn_kernel, dim3(128), gblk, 0, stream,
                     qh, kh, vT, bias, oh);

  // 5) output projection, f32 result
  hipLaunchKernelGGL(nmha_gemm_kernel, dim3(256), gblk, 0, stream,
                     oh, WTo, bo, 1.0f, 2, (__bf16*)nullptr, out);
}
